// MultiHeadAttention_67963562492254
// MI455X (gfx1250) — compile-verified, matrix operand order checked by reading, not run on hardware
//
#include <hip/hip_runtime.h>
#include <math.h>

// MHA forward for MI455X (gfx1250, wave32, WMMA).
// B=2, S=2048, D=1024, H=16, HS=64.
#define B_   2
#define S_   2048
#define D_   1024
#define H_   16
#define HS_  64
#define NOUT 1024  // H_*HS_

typedef __bf16 bf16_t;
typedef __attribute__((ext_vector_type(16))) __bf16 v16bf;
typedef __attribute__((ext_vector_type(8)))  __bf16 v8bf;
typedef __attribute__((ext_vector_type(4)))  __bf16 v4bf;
typedef __attribute__((ext_vector_type(8)))  float  v8f;
typedef __attribute__((ext_vector_type(4)))  float  v4f;

// ---------------------------------------------------------------------------
// Async global->LDS copy (CDNA5 GLOBAL_LOAD_ASYNC_TO_LDS_B128, ASYNCcnt).
// Probe round 2 revealed the builtin's param type: (global int4*, local int4*,
// imm offset, imm cpol).  Guarded: falls back to load+ds_store if absent.
// ---------------------------------------------------------------------------
#if __has_builtin(__builtin_amdgcn_global_load_async_to_lds_b128)
#define ASYNC_COPY 1
typedef int v4i_gcc __attribute__((vector_size(16)));
typedef __attribute__((address_space(1))) v4i_gcc as1_v4i;
typedef __attribute__((address_space(3))) v4i_gcc as3_v4i;
static __device__ __forceinline__ void cp16(void* lds_dst, const void* gsrc) {
  __builtin_amdgcn_global_load_async_to_lds_b128(
      (as1_v4i*)gsrc, (as3_v4i*)lds_dst, 0, 0);
}
#if __has_builtin(__builtin_amdgcn_s_wait_asynccnt)
#define WAIT_ASYNC(n) __builtin_amdgcn_s_wait_asynccnt(n)
#else
#define WAIT_ASYNC(n) asm volatile("s_wait_asynccnt %0" ::"i"(n) : "memory")
#endif
#else
#define ASYNC_COPY 0
static __device__ __forceinline__ void cp16(void* lds_dst, const void* gsrc) {
  *(v8bf*)lds_dst = *(const v8bf*)gsrc;  // global_load_b128 + ds_store_b128
}
#define WAIT_ASYNC(n) ((void)0)
#endif

// Per-lane WMMA 16-bit operand fragment (A: lane=M row, B(n-major): lane=N col).
// Per CDNA5 ISA layout each lane holds two contiguous 8-element (16B) chunks:
// K = [h*8, h*8+8) and [16+h*8, 16+h*8+8).  p already includes the h*8 offset.
static __device__ __forceinline__ v16bf load_frag(const bf16_t* p) {
  v8bf lo = *(const v8bf*)p;
  v8bf hi = *(const v8bf*)(p + 16);
  return __builtin_shufflevector(lo, hi, 0,1,2,3,4,5,6,7,8,9,10,11,12,13,14,15);
}

// ---------------------------------------------------------------- prep kernels
__global__ void cvt_x_kernel(const float* __restrict__ x, bf16_t* __restrict__ xb, int n) {
  int i = (blockIdx.x * blockDim.x + threadIdx.x) * 4;
  if (i < n) {
    float4 v = *(const float4*)(x + i);
    v4bf o = { (bf16_t)v.x, (bf16_t)v.y, (bf16_t)v.z, (bf16_t)v.w };
    *(v4bf*)(xb + i) = o;
  }
}

// Wt[z][n][d] = (bf16) W_z[d][n]   (n-major bf16 weights for WMMA B operands)
__global__ void wtrans_kernel(const float* __restrict__ WQ, const float* __restrict__ WK,
                              const float* __restrict__ WV, bf16_t* __restrict__ Wt3) {
  __shared__ float tile[32][33];
  const float* W = (blockIdx.z == 0) ? WQ : (blockIdx.z == 1) ? WK : WV;
  bf16_t* dst = Wt3 + (size_t)blockIdx.z * D_ * NOUT;
  int d0 = blockIdx.x * 32;
  int n0 = blockIdx.y * 32;
  for (int r = threadIdx.y; r < 32; r += 8)
    tile[r][threadIdx.x] = W[(size_t)(d0 + r) * NOUT + n0 + threadIdx.x];
  __syncthreads();
  for (int r = threadIdx.y; r < 32; r += 8)
    dst[(size_t)(n0 + r) * D_ + d0 + threadIdx.x] = (bf16_t)tile[threadIdx.x][r];
}

// ------------------------------------------------------------- QKV projection
// grid(32,16,3) block(256). WG tile 128(M)x64(N); K-step 32, double-buffered
// LDS staging (async copies). Wave computes a 16x64 slice (4 accumulators).
// z=0/1 -> Q/K stored [b,h,s,hs]; z=2 -> V stored transposed [b,h,hs,s].
#define APAD 40  // 128 rows x 32 k (+8 pad) : 80B row stride, conflict-free b128
#define BPAD 40

__global__ void qkv_proj_kernel(const bf16_t* __restrict__ Xb,
                                const bf16_t* __restrict__ Wt3,
                                const float* __restrict__ bQ, const float* __restrict__ bK,
                                const float* __restrict__ bV,
                                bf16_t* __restrict__ Qb, bf16_t* __restrict__ Kb,
                                bf16_t* __restrict__ Vt) {
  __shared__ __align__(16) bf16_t lA[2][128 * APAD];
  __shared__ __align__(16) bf16_t lB[2][64 * BPAD];

  const int tid = threadIdx.x;
  const int wave = tid >> 5;
  const int lane = tid & 31;
  const int lm = lane & 15, lh = lane >> 4;
  const int z = blockIdx.z;
  const int mwg = blockIdx.x * 128;
  const int nbase = blockIdx.y * 64;
  const bf16_t* Wt = Wt3 + (size_t)z * D_ * NOUT;
  const float* bias = (z == 0) ? bQ : (z == 1) ? bK : bV;

  // Stage one k-slab (3 async b128 per thread: 2 for A, 1 for B).
  auto stage = [&](int buf, int d0) {
#pragma unroll
    for (int i = 0; i < 2; ++i) {
      int c = tid + i * 256;            // 512 chunks: 128 rows x 4 chunks of 8
      int row = c >> 2, ch = c & 3;
      cp16(&lA[buf][row * APAD + ch * 8], Xb + (size_t)(mwg + row) * D_ + d0 + ch * 8);
    }
    {
      int row = tid >> 2, ch = tid & 3;  // 256 chunks: 64 rows x 4 chunks of 8
      cp16(&lB[buf][row * BPAD + ch * 8], Wt + (size_t)(nbase + row) * D_ + d0 + ch * 8);
    }
  };

  v8f acc[4] = {};
  stage(0, 0);
  for (int d = 0; d < D_; d += 32) {
    int buf = (d >> 5) & 1;
    if (d + 32 < D_) {
      stage(buf ^ 1, d + 32);   // prefetch next slab into other buffer
      WAIT_ASYNC(3);            // wait for previous slab only (3 newer in flight)
    } else {
      WAIT_ASYNC(0);
    }
    __syncthreads();
    v16bf a = load_frag(&lA[buf][(wave * 16 + lm) * APAD + lh * 8]);
#pragma unroll
    for (int t = 0; t < 4; ++t) {
      v16bf bfrag = load_frag(&lB[buf][(t * 16 + lm) * BPAD + lh * 8]);
      acc[t] = __builtin_amdgcn_wmma_f32_16x16x32_bf16(false, a, false, bfrag,
                                                       (short)0, acc[t], false, false);
    }
    __syncthreads();  // all waves done reading buf before it is restaged
  }

#pragma unroll
  for (int t = 0; t < 4; ++t) {
    int c = nbase + t * 16 + lm;   // output column for this lane (C layout: lane=N)
    float bv = bias[c];
    int h = c >> 6, hs = c & 63;
    if (z < 2) {
      bf16_t* dst = (z == 0) ? Qb : Kb;
#pragma unroll
      for (int v = 0; v < 8; ++v) {
        int mrow = mwg + wave * 16 + lh * 8 + v;   // C layout: lanes16-31 hold M=8..15
        int bb = mrow >> 11, s = mrow & (S_ - 1);
        dst[(((size_t)(bb * H_ + h) * S_ + s) * HS_) + hs] = (bf16_t)(acc[t][v] + bv);
      }
    } else {
      int mrow0 = mwg + wave * 16 + lh * 8;
      int bb = mrow0 >> 11, s0 = mrow0 & (S_ - 1);
      v8bf pack;
#pragma unroll
      for (int v = 0; v < 8; ++v) pack[v] = (bf16_t)(acc[t][v] + bv);
      *(v8bf*)(Vt + (((size_t)(bb * H_ + h) * HS_ + hs) * S_ + s0)) = pack;
    }
  }
}

// ---------------------------------------------------------- flash attention
// grid(S/128, H, B) block(256). Wave owns 16 queries; all 8 waves share each
// 32-key block, so K/V tiles are staged once per WG in double-buffered LDS.
// Computes S^T = K·Q^T so q lands on the lane dim: softmax stats are
// lane-local (one shfl_xor(16)) and exp(S^T) packs in-lane into the B
// fragment of O^T += V^T · P^T.
#define KPAD 72  // 32 rows x 64 d (+8 pad)
#define VPAD 40  // 64 rows x 32 kp (+8 pad)

__global__ void attn_kernel(const bf16_t* __restrict__ Qb, const bf16_t* __restrict__ Kb,
                            const bf16_t* __restrict__ Vt, float* __restrict__ out) {
  __shared__ __align__(16) bf16_t lK[2][32 * KPAD];
  __shared__ __align__(16) bf16_t lV[2][64 * VPAD];

  const int tid = threadIdx.x;
  const int wave = tid >> 5;
  const int lane = tid & 31;
  const int lm = lane & 15, lh = lane >> 4;
  const int h = blockIdx.y, b = blockIdx.z;
  const int qbase = blockIdx.x * 128 + wave * 16;
  const int q = qbase + lm;

  const bf16_t* Qh = Qb + (size_t)(b * H_ + h) * S_ * HS_;
  const bf16_t* Kh = Kb + (size_t)(b * H_ + h) * S_ * HS_;
  const bf16_t* Vh = Vt + (size_t)(b * H_ + h) * HS_ * S_;

  // Stage one 32-key block (2 async b128 per thread).
  auto stage = [&](int buf, int kb) {
    {
      int row = tid >> 3, ch = tid & 7;  // K: 32 rows x 8 chunks of 8 (64 d)
      cp16(&lK[buf][row * KPAD + ch * 8], Kh + (size_t)(kb + row) * HS_ + ch * 8);
    }
    {
      int row = tid >> 2, ch = tid & 3;  // V^T: 64 hs rows x 4 chunks of 8 (32 kp)
      cp16(&lV[buf][row * VPAD + ch * 8], Vh + (size_t)row * S_ + kb + ch * 8);
    }
  };

  // Q as B operand (n-major == row-major [s][hs]); reused across all key blocks.
  v16bf qb0 = load_frag(Qh + (size_t)q * HS_ + lh * 8);       // d 0..31
  v16bf qb1 = load_frag(Qh + (size_t)q * HS_ + 32 + lh * 8);  // d 32..63

  v8f O[4] = {};
  float m = -1e30f, l = 0.f;

  stage(0, 0);
  for (int kb = 0; kb < S_; kb += 32) {
    int buf = (kb >> 5) & 1;
    if (kb + 32 < S_) {
      stage(buf ^ 1, kb + 32);
      WAIT_ASYNC(2);
    } else {
      WAIT_ASYNC(0);
    }
    __syncthreads();

    const bf16_t* k0 = &lK[buf][lm * KPAD + lh * 8];         // keypos kb..+15
    const bf16_t* k1 = &lK[buf][(16 + lm) * KPAD + lh * 8];  // keypos kb+16..+31
    v8f s1 = {}, s2 = {};
    s1 = __builtin_amdgcn_wmma_f32_16x16x32_bf16(false, load_frag(k0),      false, qb0, (short)0, s1, false, false);
    s1 = __builtin_amdgcn_wmma_f32_16x16x32_bf16(false, load_frag(k0 + 32), false, qb1, (short)0, s1, false, false);
    s2 = __builtin_amdgcn_wmma_f32_16x16x32_bf16(false, load_frag(k1),      false, qb0, (short)0, s2, false, false);
    s2 = __builtin_amdgcn_wmma_f32_16x16x32_bf16(false, load_frag(k1 + 32), false, qb1, (short)0, s2, false, false);

    float bm = -1e30f;
#pragma unroll
    for (int i = 0; i < 8; ++i) {
      s1[i] *= 0.125f; s2[i] *= 0.125f;            // 1/sqrt(HS)
      bm = fmaxf(bm, fmaxf(s1[i], s2[i]));
    }
    bm = fmaxf(bm, __shfl_xor(bm, 16, 32));        // combine keypos halves
    float mnew = fmaxf(m, bm);
    float alpha = __expf(m - mnew);
    float rs = 0.f;
    v16bf pb;                                       // P^T as B operand, in-lane pack
#pragma unroll
    for (int i = 0; i < 8; ++i) {
      float p1 = __expf(s1[i] - mnew);
      float p2 = __expf(s2[i] - mnew);
      rs += p1 + p2;
      pb[i] = (bf16_t)p1;        // k = lh*8 + i
      pb[8 + i] = (bf16_t)p2;    // k = 16 + lh*8 + i
    }
    rs += __shfl_xor(rs, 16, 32);
    l = l * alpha + rs;
    m = mnew;

#pragma unroll
    for (int t = 0; t < 4; ++t) {
#pragma unroll
      for (int i = 0; i < 8; ++i) O[t][i] *= alpha;          // per-q (lane) rescale
      v16bf va = load_frag(&lV[buf][(t * 16 + lm) * VPAD + lh * 8]);
      O[t] = __builtin_amdgcn_wmma_f32_16x16x32_bf16(false, va, false, pb,
                                                     (short)0, O[t], false, false);
    }
    __syncthreads();
  }

  float inv = 1.f / l;
  float* orow = out + (size_t)(b * S_ + q) * NOUT + h * HS_;
#pragma unroll
  for (int t = 0; t < 4; ++t) {
    int hs0 = t * 16 + lh * 8;                     // O^T: lane=q, vgpr=hs (contiguous)
    v4f a0 = { O[t][0] * inv, O[t][1] * inv, O[t][2] * inv, O[t][3] * inv };
    v4f a1 = { O[t][4] * inv, O[t][5] * inv, O[t][6] * inv, O[t][7] * inv };
    *(v4f*)(orow + hs0)     = a0;
    *(v4f*)(orow + hs0 + 4) = a1;
  }
}

// ------------------------------------------------------------------- launcher
extern "C" void kernel_launch(void* const* d_in, const int* in_sizes, int n_in,
                              void* d_out, int out_size, void* d_ws, size_t ws_size,
                              hipStream_t stream) {
  (void)in_sizes; (void)n_in; (void)out_size; (void)ws_size;
  const float* x  = (const float*)d_in[0];
  const float* WQ = (const float*)d_in[1];
  const float* bQ = (const float*)d_in[2];
  const float* WK = (const float*)d_in[3];
  const float* bK = (const float*)d_in[4];
  const float* WV = (const float*)d_in[5];
  const float* bV = (const float*)d_in[6];
  float* out = (float*)d_out;

  char* ws = (char*)d_ws;
  size_t off = 0;
  auto bump = [&](size_t bytes) {
    void* p = ws + off;
    off = (off + bytes + 255) & ~(size_t)255;
    return p;
  };
  bf16_t* Xb  = (bf16_t*)bump((size_t)B_ * S_ * D_ * 2);        //  8 MB
  bf16_t* Wt3 = (bf16_t*)bump((size_t)3 * D_ * NOUT * 2);       //  6 MB
  bf16_t* Qb  = (bf16_t*)bump((size_t)B_ * H_ * S_ * HS_ * 2);  //  8 MB
  bf16_t* Kb  = (bf16_t*)bump((size_t)B_ * H_ * S_ * HS_ * 2);  //  8 MB
  bf16_t* Vt  = (bf16_t*)bump((size_t)B_ * H_ * HS_ * S_ * 2);  //  8 MB

  cvt_x_kernel<<<(B_ * S_ * D_ / 4 + 255) / 256, 256, 0, stream>>>(x, Xb, B_ * S_ * D_);
  wtrans_kernel<<<dim3(D_ / 32, NOUT / 32, 3), dim3(32, 8), 0, stream>>>(WQ, WK, WV, Wt3);
  qkv_proj_kernel<<<dim3((B_ * S_) / 128, NOUT / 64, 3), dim3(256), 0, stream>>>(
      Xb, Wt3, bQ, bK, bV, Qb, Kb, Vt);
  attn_kernel<<<dim3(S_ / 128, H_, B_), dim3(256), 0, stream>>>(Qb, Kb, Vt, out);
}
